// VectorQuantizerMultiHead_77567109365982
// MI455X (gfx1250) — matrix operand order repaired and checked by reading, not compiled, not run on hardware
//
#include <hip/hip_runtime.h>
#include <hip/hip_bf16.h>

#define H  4
#define K  8192
#define D  1024
#define DH 256
#define B  8192

#define NTILES     (K / 16)
#define ROW_BYTES  528                    // 512 data + 16 pad (TDM pad_amount=3)
#define ROW_ELEMS  (ROW_BYTES / 2)        // 264 ushorts
#define TILE_BYTES (16 * ROW_BYTES)       // 8448 per (hi|lo) tile
#define TILE_ELEMS (TILE_BYTES / 2)       // 4224
#define BUF_BYTES  (2 * TILE_BYTES)       // hi + lo
#define BUF_ELEMS  (BUF_BYTES / 2)

typedef __bf16 v16bf __attribute__((ext_vector_type(16)));
typedef float  v8f   __attribute__((ext_vector_type(8)));
typedef unsigned int v4u __attribute__((ext_vector_type(4)));
typedef int v8i __attribute__((ext_vector_type(8)));
typedef int v4i __attribute__((ext_vector_type(4)));

static __device__ __forceinline__ unsigned short bf16_rne(float f) {
  union { float f; unsigned u; } c; c.f = f;
  unsigned u = c.u;
  unsigned r = (u + 0x7FFFu + ((u >> 16) & 1u)) >> 16;
  return (unsigned short)r;
}

// Split f32 arrays into (hi, lo) bf16 pairs: hi = bf16(v), lo = bf16(v - hi).
__global__ void vq_convert_split(const float* __restrict__ x,
                                 const float* __restrict__ cb,
                                 unsigned short* __restrict__ xh,
                                 unsigned short* __restrict__ xl,
                                 unsigned short* __restrict__ ch,
                                 unsigned short* __restrict__ cl) {
  const int NX = B * D;          // == H*K*DH
  const int stride = gridDim.x * blockDim.x;
  for (int i = blockIdx.x * blockDim.x + threadIdx.x; i < NX; i += stride) {
    float v = x[i];
    unsigned short hi = bf16_rne(v);
    union { unsigned u; float f; } hf; hf.u = ((unsigned)hi) << 16;
    xh[i] = hi;
    xl[i] = bf16_rne(v - hf.f);
  }
  const int NC = H * K * DH;
  for (int i = blockIdx.x * blockDim.x + threadIdx.x; i < NC; i += stride) {
    float v = cb[i];
    unsigned short hi = bf16_rne(v);
    union { unsigned u; float f; } hf; hf.u = ((unsigned)hi) << 16;
    ch[i] = hi;
    cl[i] = bf16_rne(v - hf.f);
  }
}

// ||e||^2 per (head, code): one wave per code row, f32 exact.
__global__ void vq_e2(const float* __restrict__ cb, float* __restrict__ e2) {
  const int wave = (int)((blockIdx.x * blockDim.x + threadIdx.x) >> 5);
  const int lane = threadIdx.x & 31;
  if (wave >= H * K) return;
  const float* p = cb + (size_t)wave * DH;
  float s = 0.f;
  #pragma unroll
  for (int j = 0; j < DH; j += 32) { float v = p[j + lane]; s += v * v; }
  #pragma unroll
  for (int m = 16; m; m >>= 1) s += __shfl_xor(s, m);
  if (lane == 0) e2[wave] = s;
}

// Issue one TDM load: 16 codebook rows x 256 bf16 (row-major) -> LDS,
// with 16B of LDS padding after every 512B row (bank-conflict-free reads).
// D# per CDNA5 ISA.  (6-arg builtin: clang-23 variant.)
static __device__ __forceinline__ void tdm_load_tile(unsigned long long gaddr,
                                                     unsigned lds_byte) {
  v4u g0;
  g0.x = 1u;                                              // count=1, user mode
  g0.y = lds_byte;                                        // lds_addr (bytes)
  g0.z = (unsigned)gaddr;                                 // global_addr[31:0]
  g0.w = (unsigned)((gaddr >> 32) & 0x01FFFFFFull)        // global_addr[56:32]
       | (2u << 30);                                      // type=2 ("image")
  v8i g1;
  g1[0] = (int)((1u << 16)          // data_size = 2 bytes
              | (1u << 20)          // pad_enable
              | (6u << 22)          // pad_interval: 128 DWORDs = one 512B row
              | (3u << 25));        // pad_amount:   4 DWORDs  = 16B
  g1[1] = (int)(DH << 16);          // tensor_dim0 = 256 elements
  g1[2] = (int)(K  << 16);          // tensor_dim1 = 8192 rows
  g1[3] = (int)(DH << 16);          // tile_dim0 = 256
  g1[4] = 16;                       // tile_dim1 = 16, tile_dim2 = 0
  g1[5] = DH;                       // tensor_dim0_stride = 256 elements
  g1[6] = 0;
  g1[7] = 0;
  v4i z4 = {0, 0, 0, 0};            // groups 2/3 unused (2-D tile)
  v8i z8 = {0, 0, 0, 0, 0, 0, 0, 0};
  __builtin_amdgcn_tensor_load_to_lds(g0, g1, z4, z4, z8, 0);
}

// Fused score-GEMM (bf16x3 split, f32 accumulate) + running argmin over K.
// One wave per 16-row tile; 8 waves/block share TDM-staged codebook tiles.
__global__ void __launch_bounds__(256, 1)
vq_argmin_wmma(const unsigned short* __restrict__ xh,
               const unsigned short* __restrict__ xl,
               const unsigned short* __restrict__ ch,
               const unsigned short* __restrict__ cl,
               const float* __restrict__ e2,
               int* __restrict__ codes) {
  __shared__ unsigned short smem[4 * BUF_ELEMS];   // quad-buffered hi|lo tiles

  const int lane = threadIdx.x & 31;
  const int wave = threadIdx.x >> 5;
  const int h    = blockIdx.y;
  const int m0   = blockIdx.x * 128 + wave * 16;   // first batch row of tile
  const int r    = lane & 15;
  const int half = lane >> 4;

  // ---- Preload A tile (16 rows x 256) hi/lo into registers.
  v16bf Ah[8], Al[8];
  {
    const unsigned short* rh = xh + (size_t)(m0 + r) * D + h * DH;
    const unsigned short* rl = xl + (size_t)(m0 + r) * D + h * DH;
    #pragma unroll
    for (int c = 0; c < 8; ++c) {
      const int kb = c * 32 + half * 16;
      __builtin_memcpy(&Ah[c], rh + kb, 16);
      __builtin_memcpy(((char*)&Ah[c]) + 16, rh + kb + 8, 16);
      __builtin_memcpy(&Al[c], rl + kb, 16);
      __builtin_memcpy(((char*)&Al[c]) + 16, rl + kb + 8, 16);
    }
  }

  float minv[8];
  int   mini[8];
  #pragma unroll
  for (int i = 0; i < 8; ++i) { minv[i] = 3.4e38f; mini[i] = 0; }

  const size_t cb_head = (size_t)h * K * DH;
  const float* e2h = e2 + (size_t)h * K;
  const unsigned lds0 = (unsigned)(unsigned long long)(uintptr_t)&smem[0];
  const unsigned long long ghb = (unsigned long long)(uintptr_t)(ch + cb_head);
  const unsigned long long glb = (unsigned long long)(uintptr_t)(cl + cb_head);

  // Prefetch tiles 0 and 1 (wave 0 drives the Tensor Data Mover).
  if (wave == 0) {
    tdm_load_tile(ghb, lds0);
    tdm_load_tile(glb, lds0 + TILE_BYTES);
    tdm_load_tile(ghb + BUF_BYTES / 2, lds0 + BUF_BYTES);
    tdm_load_tile(glb + BUF_BYTES / 2, lds0 + BUF_BYTES + TILE_BYTES);
  }

  for (int nt = 0; nt < NTILES; ++nt) {
    if (wave == 0) {
      if (nt + 2 < NTILES) {
        const int t = nt + 2;
        const unsigned long long goff = (unsigned long long)t * (16 * DH * 2);
        const unsigned lb = lds0 + (unsigned)(t & 3) * BUF_BYTES;
        tdm_load_tile(ghb + goff, lb);
        tdm_load_tile(glb + goff, lb + TILE_BYTES);
        // 2 tiles (4 TDM ops) may stay in flight; oldest tile retired.
        __builtin_amdgcn_s_wait_tensorcnt(4);
      } else {
        __builtin_amdgcn_s_wait_tensorcnt(0);
      }
    }
    // One barrier/iter: quad buffering guarantees the buffer written above
    // was consumed two iterations ago (separated by the previous barrier).
    __syncthreads();

    const unsigned short* th = smem + (nt & 3) * BUF_ELEMS;
    const unsigned short* tl = th + TILE_ELEMS;
    const int n = nt * 16 + r;            // code column this lane covers
    const int rowoff = r * ROW_ELEMS;     // padded row stride: no bank conflicts

    v8f acc = {};
    #pragma unroll
    for (int c = 0; c < 8; ++c) {
      const int kb = c * 32 + half * 16;
      v16bf Bh, Bl;
      __builtin_memcpy(&Bh, th + rowoff + kb, 32);
      __builtin_memcpy(&Bl, tl + rowoff + kb, 32);
      acc = __builtin_amdgcn_wmma_f32_16x16x32_bf16(false, Ah[c], false, Bh,
                                                    (short)0, acc, false, false);
      acc = __builtin_amdgcn_wmma_f32_16x16x32_bf16(false, Ah[c], false, Bl,
                                                    (short)0, acc, false, false);
      acc = __builtin_amdgcn_wmma_f32_16x16x32_bf16(false, Al[c], false, Bh,
                                                    (short)0, acc, false, false);
    }

    // dist (up to the row-constant ||x||^2) = ||e||^2 - 2 * <x,e>
    const float e2v = e2h[n];
    #pragma unroll
    for (int i = 0; i < 8; ++i) {
      const float dist = e2v - 2.0f * acc[i];
      if (dist < minv[i]) { minv[i] = dist; mini[i] = n; }  // strict < keeps first
    }
  }

  // Cross-lane argmin within each 16-lane half; tie-break to lowest index.
  #pragma unroll
  for (int i = 0; i < 8; ++i) {
    float v = minv[i]; int idx = mini[i];
    #pragma unroll
    for (int m = 8; m; m >>= 1) {
      const float ov = __shfl_xor(v, m);
      const int   oi = __shfl_xor(idx, m);
      if (ov < v || (ov == v && oi < idx)) { v = ov; idx = oi; }
    }
    minv[i] = v; mini[i] = idx;
  }
  if (r == 0) {
    #pragma unroll
    for (int i = 0; i < 8; ++i) {
      const int b = m0 + half * 8 + i;   // VGPR i: rows i (lanes 0-15), i+8 (16-31)
      codes[b * H + h] = mini[i];
    }
  }
}

// Exact f32 epilogue: gather selected codebook rows -> quantized; loss.
__global__ void vq_epilogue(const float* __restrict__ x,
                            const float* __restrict__ cb,
                            const int* __restrict__ codes,
                            float* __restrict__ loss,
                            float* __restrict__ quant) {
  __shared__ float s[4];
  const int b    = blockIdx.x;
  const int w    = threadIdx.x >> 5;   // head
  const int lane = threadIdx.x & 31;
  const int code = codes[b * H + w];
  const float* xp = x  + (size_t)b * D + w * DH;
  const float* qp = cb + ((size_t)w * K + code) * DH;
  float* op = quant + (size_t)b * D + w * DH;
  float sum = 0.f;
  #pragma unroll
  for (int j = 0; j < DH; j += 32) {
    const float q  = qp[j + lane];
    const float xv = xp[j + lane];
    op[j + lane] = q;                  // straight-through output == q numerically
    const float df = q - xv;
    sum += df * df;
  }
  #pragma unroll
  for (int m = 16; m; m >>= 1) sum += __shfl_xor(sum, m);
  if (lane == 0) s[w] = sum;
  __syncthreads();
  if (threadIdx.x == 0)
    loss[b] = 0.25f * (s[0] + s[1] + s[2] + s[3]) * (1.0f / DH);
}

extern "C" void kernel_launch(void* const* d_in, const int* in_sizes, int n_in,
                              void* d_out, int out_size, void* d_ws, size_t ws_size,
                              hipStream_t stream) {
  const float* x  = (const float*)d_in[0];   // (B, 1, D) f32
  const float* cb = (const float*)d_in[1];   // (H, K, d) f32

  float* out   = (float*)d_out;
  float* loss  = out;                                  // (B,)
  float* quant = out + B;                              // (B, 1, D)
  int*   codes = (int*)(out + B + (size_t)B * D);      // (B, H) int32

  const size_t NX = (size_t)B * D;
  const size_t NC = (size_t)H * K * DH;
  unsigned short* xh = (unsigned short*)d_ws;
  unsigned short* xl = xh + NX;
  unsigned short* ch = xl + NX;
  unsigned short* cl = ch + NC;
  float*          e2 = (float*)(cl + NC);              // H*K f32

  vq_convert_split<<<2048, 256, 0, stream>>>(x, cb, xh, xl, ch, cl);
  vq_e2<<<(H * K) / 8, 256, 0, stream>>>(cb, e2);

  dim3 grid(B / 128, H);
  vq_argmin_wmma<<<grid, 256, 0, stream>>>(xh, xl, ch, cl, e2, codes);

  vq_epilogue<<<B, 128, 0, stream>>>(x, cb, codes, loss, quant);
}
